// WindowAttention_51917564674156
// MI455X (gfx1250) — compile-verified, matrix operand order checked by reading, not run on hardware
//
#include <hip/hip_runtime.h>

#define EMB   192
#define HEADS 6
#define NTOK  64
#define HDIM  32
#define BWIN  2048
#define NWIN  1024

typedef __attribute__((ext_vector_type(16))) __bf16 v16bf;
typedef __attribute__((ext_vector_type(8)))  float  v8f;

// Load a 16x32 bf16 WMMA fragment where each lane holds one row (A usage) or
// one output-column-as-row (B usage, matrix pre-transposed so its rows are the
// output columns). CDNA5 16-bit operand layout:
//   lane<16 : K = {0..7, 16..23},  lane>=16 : K = {8..15, 24..31}
// which is contiguous-per-half -> two 16B loads.
__device__ __forceinline__ v16bf load_frag(const __bf16* base, int row0, int ldm,
                                           int k0, int lane) {
  const int hf = lane >> 4, r = lane & 15;
  const __bf16* p = base + (row0 + r) * ldm + k0 + (hf << 3);
  union { v16bf v; uint4 u[2]; } t;
  t.u[0] = *(const uint4*)p;
  t.u[1] = *(const uint4*)(p + 16);
  return t.v;
}

__device__ __forceinline__ unsigned int pack_bf16(float a, float b) {
  union { __bf16 h[2]; unsigned int u; } t;
  t.h[0] = (__bf16)a; t.h[1] = (__bf16)b;
  return t.u;
}

// Prep: gather relative-position bias per head, and transpose+convert the four
// 192x192 weight matrices to bf16 column-major (rows = output channels).
__global__ void __launch_bounds__(256) prep_kernel(
    const float* __restrict__ bias_table, const int* __restrict__ rel_index,
    const float* __restrict__ Wq, const float* __restrict__ Wk,
    const float* __restrict__ Wv, const float* __restrict__ Wp,
    float* __restrict__ rel_bias, __bf16* __restrict__ Wt) {
  int i = blockIdx.x * 256 + threadIdx.x;
  if (i < HEADS * NTOK * NTOK) {
    int h = i / (NTOK * NTOK), nm = i % (NTOK * NTOK);
    rel_bias[i] = bias_table[rel_index[nm] * HEADS + h];
  }
  for (int idx = i; idx < 4 * EMB * EMB; idx += gridDim.x * 256) {
    int p = idx / (EMB * EMB), r = idx % (EMB * EMB);
    int kk = r / EMB, c = r % EMB;
    const float* W = (p == 0) ? Wq : (p == 1 ? Wk : (p == 2 ? Wv : Wp));
    Wt[p * EMB * EMB + c * EMB + kk] = (__bf16)W[kk * EMB + c];
  }
}

__global__ void __launch_bounds__(256) win_attn_kernel(
    const float* __restrict__ q, const float* __restrict__ k,
    const float* __restrict__ v, const float* __restrict__ mask,
    const float* __restrict__ bq, const float* __restrict__ bk,
    const float* __restrict__ bv, const float* __restrict__ bp,
    const float* __restrict__ rel_bias, const __bf16* __restrict__ Wt,
    float* __restrict__ out_x, float* __restrict__ out_attn) {
  __shared__ __bf16 smemA[3 * NTOK * EMB];  // raw q,k,v ; later P(6*64*64) + x(64*192)
  __shared__ __bf16 smemB[3 * NTOK * EMB];  // qh (row-major), kh (row-major), vT (chan-major)

  const int b    = blockIdx.x;
  const int tid  = threadIdx.x;
  const int lane = tid & 31;
  const int wave = tid >> 5;
  const int hf   = lane >> 4;
  const int col  = lane & 15;

  // ---------- Phase 0: stage q,k,v (64x192 f32 each) -> bf16 LDS ----------
  {
    const float* srcs[3] = {q + (size_t)b * NTOK * EMB,
                            k + (size_t)b * NTOK * EMB,
                            v + (size_t)b * NTOK * EMB};
#pragma unroll 1
    for (int mtx = 0; mtx < 3; ++mtx) {
      const float4* src = (const float4*)srcs[mtx];
      __bf16* dst = smemA + mtx * NTOK * EMB;
      for (int e4 = tid; e4 < NTOK * EMB / 4; e4 += 256) {
        float4 f = src[e4];
        uint2 p;
        p.x = pack_bf16(f.x, f.y);
        p.y = pack_bf16(f.z, f.w);
        *(uint2*)(dst + e4 * 4) = p;   // ds_store_b64, cvt_pk path
      }
    }
  }
  __syncthreads();

  // ---------- Phase 1: QKV projections, 36 (proj,nt) combos over 8 waves ----------
  const float scale = 0.17677669529663689f;  // 32^-0.5 folded into qh
  for (int combo = wave; combo < 36; combo += 8) {
    int proj = combo / 12, nt = combo % 12;
    const __bf16* WtP = Wt + proj * EMB * EMB;
    v16bf Bf[6];
#pragma unroll
    for (int s = 0; s < 6; ++s) Bf[s] = load_frag(WtP, nt * 16, EMB, 32 * s, lane);
    const float* bvec = (proj == 0) ? bq : (proj == 1 ? bk : bv);
    float bias = bvec[nt * 16 + col];
    float mult = (proj == 0) ? scale : 1.0f;
    const __bf16* X = smemA + proj * NTOK * EMB;
    bool isV = (proj == 2);
    int baseRM = proj * NTOK * EMB;            // row-major (q/k)
#pragma unroll 2
    for (int mt = 0; mt < 4; ++mt) {
      v8f acc = {};
#pragma unroll
      for (int s = 0; s < 6; ++s) {
        v16bf A = load_frag(X, mt * 16, EMB, 32 * s, lane);
        acc = __builtin_amdgcn_wmma_f32_16x16x32_bf16(false, A, false, Bf[s],
                                                      (short)0, acc, false, false);
      }
      int c = nt * 16 + col;
#pragma unroll
      for (int r = 0; r < 8; ++r) {
        float val = (acc[r] + bias) * mult;
        int rowg = mt * 16 + r + hf * 8;                 // sequence position
        int addrRM = baseRM + rowg * EMB + c;            // qh/kh row-major
        int addrT  = 2 * NTOK * EMB + c * NTOK + rowg;   // vT chan-major
        smemB[isV ? addrT : addrRM] = (__bf16)val;       // branch-free select
      }
    }
  }
  __syncthreads();

  // ---------- Phase 2: S = qh*kh^T + bias + mask, softmax -> attn out + P in LDS ----------
  const __bf16* qh = smemB;
  const __bf16* kh = smemB + NTOK * EMB;
  const __bf16* vT = smemB + 2 * NTOK * EMB;
  __bf16* Psm = smemA;                            // [h][n][m] bf16
  __bf16* Xsm = smemA + HEADS * NTOK * NTOK;      // [n][c]   bf16

  const int wid = b & (NWIN - 1);
  const float* mrow = mask + (size_t)wid * NTOK * NTOK;

  for (int strip = wave; strip < 24; strip += 8) {  // strip = (head, 16-row block)
    int h = strip >> 2, nt = strip & 3;
    v16bf A = load_frag(qh, nt * 16, EMB, h * HDIM, lane);
    v8f S[4];
#pragma unroll
    for (int mt = 0; mt < 4; ++mt) {
      v16bf Bf = load_frag(kh, mt * 16, EMB, h * HDIM, lane);
      v8f z = {};
      S[mt] = __builtin_amdgcn_wmma_f32_16x16x32_bf16(false, A, false, Bf,
                                                      (short)0, z, false, false);
    }
#pragma unroll
    for (int mt = 0; mt < 4; ++mt)
#pragma unroll
      for (int r = 0; r < 8; ++r) {
        int n = nt * 16 + r + hf * 8;
        int m = mt * 16 + col;
        S[mt][r] += rel_bias[(h * NTOK + n) * NTOK + m] + mrow[n * NTOK + m];
      }
    // Row softmax: row n lives across 16 lanes of one half; reduce with width-16 xor shuffles.
    float mx[8], inv[8];
#pragma unroll
    for (int r = 0; r < 8; ++r) {
      float m0 = fmaxf(fmaxf(S[0][r], S[1][r]), fmaxf(S[2][r], S[3][r]));
      m0 = fmaxf(m0, __shfl_xor(m0, 1, 16));
      m0 = fmaxf(m0, __shfl_xor(m0, 2, 16));
      m0 = fmaxf(m0, __shfl_xor(m0, 4, 16));
      m0 = fmaxf(m0, __shfl_xor(m0, 8, 16));
      mx[r] = m0;
    }
#pragma unroll
    for (int mt = 0; mt < 4; ++mt)
#pragma unroll
      for (int r = 0; r < 8; ++r) S[mt][r] = __expf(S[mt][r] - mx[r]);
#pragma unroll
    for (int r = 0; r < 8; ++r) {
      float s0 = S[0][r] + S[1][r] + S[2][r] + S[3][r];
      s0 += __shfl_xor(s0, 1, 16);
      s0 += __shfl_xor(s0, 2, 16);
      s0 += __shfl_xor(s0, 4, 16);
      s0 += __shfl_xor(s0, 8, 16);
      inv[r] = 1.0f / s0;
    }
    float* outA = out_attn + ((size_t)(b * HEADS + h)) * NTOK * NTOK;
#pragma unroll
    for (int mt = 0; mt < 4; ++mt)
#pragma unroll
      for (int r = 0; r < 8; ++r) {
        int n = nt * 16 + r + hf * 8;
        int m = mt * 16 + col;
        float p = S[mt][r] * inv[r];
        outA[n * NTOK + m] = p;
        Psm[(h * NTOK + n) * NTOK + m] = (__bf16)p;
      }
  }
  __syncthreads();

  // ---------- Phase 3: x = P * V  (48 16x16 tiles over 8 waves) ----------
  for (int t = wave; t < 48; t += 8) {
    int h = t >> 3, rem = t & 7, nt = rem >> 1, dt = rem & 1;
    v8f acc = {};
#pragma unroll
    for (int kk = 0; kk < 2; ++kk) {
      v16bf A  = load_frag(Psm + h * NTOK * NTOK, nt * 16, NTOK, kk * 32, lane);
      v16bf Bf = load_frag(vT, h * HDIM + dt * 16, NTOK, kk * 32, lane);  // vT rows=channels
      acc = __builtin_amdgcn_wmma_f32_16x16x32_bf16(false, A, false, Bf,
                                                    (short)0, acc, false, false);
    }
#pragma unroll
    for (int r = 0; r < 8; ++r) {
      int n = nt * 16 + r + hf * 8;
      int c = h * HDIM + dt * 16 + col;
      Xsm[n * EMB + c] = (__bf16)acc[r];
    }
  }
  __syncthreads();

  // ---------- Phase 4: out = x * Wp + bp ----------
  {
    const __bf16* WtP = Wt + 3 * EMB * EMB;
    float* o = out_x + (size_t)b * NTOK * EMB;
    for (int nt = wave; nt < 12; nt += 8) {
      v16bf Bf[6];
#pragma unroll
      for (int s = 0; s < 6; ++s) Bf[s] = load_frag(WtP, nt * 16, EMB, 32 * s, lane);
      float bias = bp[nt * 16 + col];
#pragma unroll 2
      for (int mt = 0; mt < 4; ++mt) {
        v8f acc = {};
#pragma unroll
        for (int s = 0; s < 6; ++s) {
          v16bf A = load_frag(Xsm, mt * 16, EMB, 32 * s, lane);
          acc = __builtin_amdgcn_wmma_f32_16x16x32_bf16(false, A, false, Bf[s],
                                                        (short)0, acc, false, false);
        }
#pragma unroll
        for (int r = 0; r < 8; ++r) {
          int n = mt * 16 + r + hf * 8;
          o[n * EMB + nt * 16 + col] = acc[r] + bias;
        }
      }
    }
  }
}

extern "C" void kernel_launch(void* const* d_in, const int* in_sizes, int n_in,
                              void* d_out, int out_size, void* d_ws, size_t ws_size,
                              hipStream_t stream) {
  const float* q          = (const float*)d_in[0];
  const float* k          = (const float*)d_in[1];
  const float* v          = (const float*)d_in[2];
  const float* mask       = (const float*)d_in[3];
  const float* Wq         = (const float*)d_in[4];
  const float* bq         = (const float*)d_in[5];
  const float* Wk         = (const float*)d_in[6];
  const float* bk         = (const float*)d_in[7];
  const float* Wv         = (const float*)d_in[8];
  const float* bv         = (const float*)d_in[9];
  const float* Wp         = (const float*)d_in[10];
  const float* bp         = (const float*)d_in[11];
  const float* bias_table = (const float*)d_in[12];
  const int*   rel_index  = (const int*)d_in[13];

  float*  rel_bias = (float*)d_ws;                                   // 6*64*64 f32
  __bf16* Wt = (__bf16*)((char*)d_ws + HEADS * NTOK * NTOK * sizeof(float)); // 4*192*192 bf16

  float* out_x    = (float*)d_out;
  float* out_attn = out_x + (size_t)BWIN * NTOK * EMB;

  prep_kernel<<<576, 256, 0, stream>>>(bias_table, rel_index, Wq, Wk, Wv, Wp,
                                       rel_bias, Wt);
  win_attn_kernel<<<BWIN, 256, 0, stream>>>(q, k, v, mask, bq, bk, bv, bp,
                                            rel_bias, Wt, out_x, out_attn);
}
